// NeoVisionGNN_30021821399627
// MI455X (gfx1250) — matrix operand order repaired and checked by reading, not compiled, not run on hardware
//
#include <hip/hip_runtime.h>
#include <hip/hip_bf16.h>
#include <cstdint>

// ---------------------------------------------------------------------------
// NeoVisionGNN on MI455X (gfx1250, wave32, WMMA)
//
// B=16, C=384, H=W=32 -> N=1024 nodes/graph, 16384 nodes total, K=9 knn.
// Gram matrix (12.9 GFLOP) -> bf16 WMMA (feeds top-k selection only).
// Linear    ( 4.8 GFLOP)   -> f32 WMMA 16x16x4 (feeds output, keep exact).
// deg == K+1 == 10 for every node -> GCN norm is a constant 0.1.
// ---------------------------------------------------------------------------

typedef __attribute__((ext_vector_type(16))) __bf16 v16bf;
typedef __attribute__((ext_vector_type(8)))  __bf16 v8bf;
typedef __attribute__((ext_vector_type(8)))  float  v8f;
typedef __attribute__((ext_vector_type(2)))  float  v2f;

#define B_    16
#define C_    384
#define N_    1024
#define ROWS_ (B_ * N_)      // 16384
#define KNN_  9

// ---------------------------------------------------------------------------
// K1: [B, C, N] -> node-major [B*N, C], fp32 + bf16 copies. LDS-tiled.
// grid (N/32, C/32, B), block (32, 8)
// ---------------------------------------------------------------------------
__global__ __launch_bounds__(256) void k_transpose(const float* __restrict__ x,
                                                   float* __restrict__ xn,
                                                   __bf16* __restrict__ xbf) {
    __shared__ float tile[32][33];
    const int b  = blockIdx.z;
    const int c0 = blockIdx.y * 32;
    const int n0 = blockIdx.x * 32;
    const int tx = threadIdx.x, ty = threadIdx.y;
#pragma unroll
    for (int i = 0; i < 4; ++i) {
        int c = c0 + ty + i * 8;
        tile[ty + i * 8][tx] = x[((size_t)b * C_ + c) * N_ + (n0 + tx)];
    }
    __syncthreads();
#pragma unroll
    for (int i = 0; i < 4; ++i) {
        int n = n0 + ty + i * 8;
        int c = c0 + tx;
        float v = tile[tx][ty + i * 8];
        size_t o = ((size_t)b * N_ + n) * C_ + c;
        xn[o]  = v;
        xbf[o] = (__bf16)v;
    }
}

// ---------------------------------------------------------------------------
// K2: per-node squared norm. One wave32 per row, 8 rows per 256-thread block.
// ---------------------------------------------------------------------------
__global__ __launch_bounds__(256) void k_norms(const float* __restrict__ xn,
                                               float* __restrict__ sq) {
    const int wave = threadIdx.x >> 5;
    const int lane = threadIdx.x & 31;
    const int row  = blockIdx.x * 8 + wave;
    const float* p = xn + (size_t)row * C_;
    float s = 0.0f;
#pragma unroll
    for (int i = lane; i < C_; i += 32) { float v = p[i]; s += v * v; }
#pragma unroll
    for (int off = 16; off > 0; off >>= 1) s += __shfl_xor(s, off, 32);
    if (lane == 0) sq[row] = s;
}

// ---------------------------------------------------------------------------
// K3: fused pairwise-distance (bf16 WMMA) + per-row top-9 selection.
// One wave per 16 query rows. grid = B * (N/16) = 1024 blocks of 32 threads.
// A-panel (16 rows x 384) preloaded in registers (12 x v16bf).
// D never materialized in global memory.
// ---------------------------------------------------------------------------
__global__ __launch_bounds__(32) void k_knn(const __bf16* __restrict__ xbf,
                                            const float* __restrict__ sq,
                                            int* __restrict__ nbr) {
    const int tile   = blockIdx.x;     // 0..1023
    const int b      = tile >> 6;      // / (N/16)
    const int qt     = tile & 63;
    const int qt16   = qt * 16;
    const int lane   = threadIdx.x;
    const int lane16 = lane & 15;
    const int hi     = lane >> 4;      // 0: lanes 0-15, 1: lanes 16-31

    const __bf16* xb  = xbf + (size_t)b * N_ * C_;
    const float*  sqb = sq  + (size_t)b * N_;

    // ---- preload A tiles for all 12 k-chunks (contiguous k order, same as B) ----
    // element e of v16bf: K = hi*16 + e  (within a 32-wide chunk)
    v16bf areg[12];
    {
        const __bf16* qrow = xb + (size_t)(qt16 + lane16) * C_;
        const int ka = hi * 16;
#pragma unroll
        for (int kc = 0; kc < 12; ++kc) {
            v8bf lo = *(const v8bf*)(qrow + kc * 32 + ka);
            v8bf hh = *(const v8bf*)(qrow + kc * 32 + ka + 8);
            v16bf t;
#pragma unroll
            for (int e = 0; e < 8; ++e) { t[e] = lo[e]; t[e + 8] = hh[e]; }
            areg[kc] = t;
        }
    }

    float sqq[8];
#pragma unroll
    for (int v = 0; v < 8; ++v) sqq[v] = sqb[qt16 + v + hi * 8];

    // per-row (lanes 0..15) sorted ascending top-9 {distance, index}
    float bd[KNN_]; int bi[KNN_];
#pragma unroll
    for (int j = 0; j < KNN_; ++j) { bd[j] = 1e30f; bi[j] = 0; }

    __shared__ float Dt[16][17];

    const int kb = hi * 16;   // B k order, same as A: element e -> K = hi*16 + e
    for (int mt = 0; mt < 64; ++mt) {
        const __bf16* mrow = xb + (size_t)(mt * 16 + lane16) * C_;
        v8f acc = {};
#pragma unroll
        for (int kc = 0; kc < 12; ++kc) {
            v16bf bt = *(const v16bf*)(mrow + kc * 32 + kb);
            acc = __builtin_amdgcn_wmma_f32_16x16x32_bf16(
                false, areg[kc], false, bt, (short)0, acc, false, false);
        }
        const float sqm = sqb[mt * 16 + lane16];
        // C/D layout: lane vgpr v holds (row = v + hi*8, col = lane16)
#pragma unroll
        for (int v = 0; v < 8; ++v) {
            const int rowm = v + hi * 8;
            float d = sqq[v] + sqm - 2.0f * acc[v];
            if (qt16 + rowm == mt * 16 + lane16) d = 1e30f;  // exclude self
            Dt[rowm][lane16] = d;
        }
        __syncthreads();
        if (lane < 16) {
#pragma unroll
            for (int col = 0; col < 16; ++col) {
                float cd = Dt[lane][col];
                int   ci = mt * 16 + col;
                if (cd < bd[KNN_ - 1]) {
#pragma unroll
                    for (int j = 0; j < KNN_; ++j) {
                        if (cd < bd[j]) {
                            float tf = bd[j]; bd[j] = cd; cd = tf;
                            int   ti = bi[j]; bi[j] = ci; ci = ti;
                        }
                    }
                }
            }
        }
        __syncthreads();
    }

    if (lane < 16) {
        size_t base = ((size_t)b * N_ + qt16 + lane) * KNN_;
#pragma unroll
        for (int j = 0; j < KNN_; ++j) nbr[base + j] = bi[j];
    }
}

// ---------------------------------------------------------------------------
// K4: xw = xn @ Wg^T in exact f32 via v_wmma_f32_16x16x4_f32.
// One wave per 16x16 output tile; grid = (16384/16) * (384/16) = 24576.
// ---------------------------------------------------------------------------
__global__ __launch_bounds__(32) void k_linear(const float* __restrict__ xn,
                                               const float* __restrict__ Wg,
                                               float* __restrict__ xw) {
    const int tile   = blockIdx.x;
    const int nt     = tile / (C_ / 16);
    const int ct     = tile % (C_ / 16);
    const int lane   = threadIdx.x;
    const int lane16 = lane & 15;
    const int hi     = lane >> 4;

    // f32 A 16x4 layout: a[j] -> (row = lane16, K = hi*2 + j)
    // f32 B  4x16 layout: b[j] -> (K = hi*2 + j, col = lane16);  B[k][n] = Wg[n][k]
    const float* arow = xn + (size_t)(nt * 16 + lane16) * C_ + hi * 2;
    const float* brow = Wg + (size_t)(ct * 16 + lane16) * C_ + hi * 2;

    v8f acc = {};
#pragma unroll 4
    for (int kc = 0; kc < C_ / 4; ++kc) {
        v2f a = *(const v2f*)(arow + kc * 4);
        v2f w = *(const v2f*)(brow + kc * 4);
        acc = __builtin_amdgcn_wmma_f32_16x16x4_f32(
            false, a, false, w, (short)0, acc, false, false);
    }
#pragma unroll
    for (int v = 0; v < 8; ++v)
        xw[(size_t)(nt * 16 + v + hi * 8) * C_ + ct * 16 + lane16] = acc[v];
}

// ---------------------------------------------------------------------------
// K5: GCN aggregation. deg == 10 everywhere -> norm = rsqrt(10)^2 constant.
// out[i] = norm * (xw[i] + sum_j xw[nbr_j]) + bg.  One block per node.
// ---------------------------------------------------------------------------
__global__ __launch_bounds__(128) void k_agg(const float* __restrict__ xw,
                                             const int* __restrict__ nbr,
                                             const float* __restrict__ bg,
                                             float* __restrict__ agg) {
    const int node = blockIdx.x;           // 0..16383
    const int b    = node >> 10;           // / N_
    __shared__ int nb[KNN_];
    if (threadIdx.x < KNN_) nb[threadIdx.x] = nbr[(size_t)node * KNN_ + threadIdx.x];
    __syncthreads();
    const float di     = rsqrtf((float)(KNN_ + 1));
    const float factor = di * di;
    for (int c = threadIdx.x; c < C_; c += 128) {
        float s = xw[(size_t)node * C_ + c];
#pragma unroll
        for (int j = 0; j < KNN_; ++j)
            s += xw[(size_t)(b * N_ + nb[j]) * C_ + c];
        agg[(size_t)node * C_ + c] = factor * s + bg[c];
    }
}

// ---------------------------------------------------------------------------
// K6: BatchNorm training-mode stats per channel (biased variance).
// One block per channel, 256 threads.
// ---------------------------------------------------------------------------
__global__ __launch_bounds__(256) void k_bnstats(const float* __restrict__ agg,
                                                 float* __restrict__ mean,
                                                 float* __restrict__ rstd) {
    const int c = blockIdx.x;
    __shared__ float s1[256], s2[256];
    float a = 0.0f, q = 0.0f;
    for (int r = threadIdx.x; r < ROWS_; r += 256) {
        float v = agg[(size_t)r * C_ + c];
        a += v; q += v * v;
    }
    s1[threadIdx.x] = a; s2[threadIdx.x] = q;
    __syncthreads();
    for (int off = 128; off > 0; off >>= 1) {
        if (threadIdx.x < off) {
            s1[threadIdx.x] += s1[threadIdx.x + off];
            s2[threadIdx.x] += s2[threadIdx.x + off];
        }
        __syncthreads();
    }
    if (threadIdx.x == 0) {
        float mu  = s1[0] / (float)ROWS_;
        float var = s2[0] / (float)ROWS_ - mu * mu;
        mean[c] = mu;
        rstd[c] = rsqrtf(var + 1e-5f);
    }
}

// ---------------------------------------------------------------------------
// K7: BN apply + tanh-GELU + residual, writing [B, C, N] output layout.
// ---------------------------------------------------------------------------
__global__ __launch_bounds__(256) void k_final(const float* __restrict__ agg,
                                               const float* __restrict__ x,
                                               const float* __restrict__ gamma,
                                               const float* __restrict__ beta,
                                               const float* __restrict__ mean,
                                               const float* __restrict__ rstd,
                                               float* __restrict__ out) {
    const size_t idx = (size_t)blockIdx.x * 256 + threadIdx.x;
    if (idx >= (size_t)B_ * C_ * N_) return;
    const int n = (int)(idx % N_);
    const int c = (int)((idx / N_) % C_);
    const int b = (int)(idx / ((size_t)N_ * C_));
    float v = agg[((size_t)b * N_ + n) * C_ + c];
    v = gamma[c] * (v - mean[c]) * rstd[c] + beta[c];
    // jax.nn.gelu default: tanh approximation
    float v3 = v * v * v;
    float g  = 0.5f * v * (1.0f + tanhf(0.7978845608028654f * (v + 0.044715f * v3)));
    out[idx] = g + x[idx];
}

// ---------------------------------------------------------------------------
// Host launch. Inputs: x, Wg, bg, gamma, beta (all fp32). Output fp32 [B,C,N].
// ---------------------------------------------------------------------------
extern "C" void kernel_launch(void* const* d_in, const int* in_sizes, int n_in,
                              void* d_out, int out_size, void* d_ws, size_t ws_size,
                              hipStream_t stream) {
    const float* x     = (const float*)d_in[0];
    const float* Wg    = (const float*)d_in[1];
    const float* bg    = (const float*)d_in[2];
    const float* gamma = (const float*)d_in[3];
    const float* beta  = (const float*)d_in[4];
    float* out = (float*)d_out;

    char* ws = (char*)d_ws;
    // all section sizes are multiples of 256 bytes
    float*  XN   = (float*) (ws);                            // 16384*384*4 = 25165824
    __bf16* XBF  = (__bf16*)(ws + 25165824);                 // 16384*384*2 = 12582912
    float*  SQ   = (float*) (ws + 37748736);                 // 16384*4     =    65536
    int*    NBR  = (int*)   (ws + 37814272);                 // 16384*9*4   =   589824
    float*  XW   = (float*) (ws + 38404096);                 // 25165824
    float*  AGG  = (float*) (ws + 63569920);                 // 25165824
    float*  MEAN = (float*) (ws + 88735744);                 // 1536
    float*  RSTD = (float*) (ws + 88737280);                 // 1536

    k_transpose<<<dim3(N_ / 32, C_ / 32, B_), dim3(32, 8), 0, stream>>>(x, XN, XBF);
    k_norms    <<<ROWS_ / 8, 256, 0, stream>>>(XN, SQ);
    k_knn      <<<B_ * (N_ / 16), 32, 0, stream>>>(XBF, SQ, NBR);
    k_linear   <<<(ROWS_ / 16) * (C_ / 16), 32, 0, stream>>>(XN, Wg, XW);
    k_agg      <<<ROWS_, 128, 0, stream>>>(XW, NBR, bg, AGG);
    k_bnstats  <<<C_, 256, 0, stream>>>(AGG, MEAN, RSTD);
    k_final    <<<(B_ * C_ * N_ + 255) / 256, 256, 0, stream>>>(AGG, x, gamma, beta,
                                                                MEAN, RSTD, out);
}
